// MultiHead_83932250898863
// MI455X (gfx1250) — matrix operand order repaired and checked by reading, not compiled, not run on hardware
//
#include <hip/hip_runtime.h>

typedef __attribute__((ext_vector_type(16))) _Float16 v16h;
typedef __attribute__((ext_vector_type(8)))  _Float16 h8;
typedef __attribute__((ext_vector_type(8)))  float    v8f;

#define B_SZ    2048
#define FLAT_SZ 102400
#define KSLICES 16
#define KPER    (FLAT_SZ / KSLICES)   // 6400
#define CHUNK   256
#define NCHUNK  (KPER / CHUNK)        // 25
#define NCHUNK_TOT (FLAT_SZ / CHUNK)  // 400
#define H1_ROWS 1696                  // 1681 padded to 106 full 16-row tiles

__device__ __forceinline__ v16h cat16(h8 lo, h8 hi) {
  return __builtin_shufflevector(lo, hi, 0,1,2,3,4,5,6,7,8,9,10,11,12,13,14,15);
}

__device__ __forceinline__ v8f wmma_f16(v16h a, v16h b, v8f c) {
  // (neg_a, A, neg_b, B, c_mod, C, reuse_a, reuse_b)
  return __builtin_amdgcn_wmma_f32_16x16x32_f16(false, a, false, b, (short)0, c, false, false);
}

// relu on the f16 side: cvt output is canonical -> single v_max_num_f16,
// and relu(cvt(x)) == cvt(relu(x)) exactly (monotone, 0-preserving).
__device__ __forceinline__ _Float16 relu_h(_Float16 v) {
#if __has_builtin(__builtin_fmaxf16)
  return __builtin_fmaxf16(v, (_Float16)0.f);
#else
  return v > (_Float16)0.f ? v : (_Float16)0.f;
#endif
}

// Async DMA: global -> LDS, 16B per lane, tracked by ASYNCcnt (no VGPR data).
__device__ __forceinline__ void async_ld_b128(unsigned lds_off, const void* gaddr) {
  asm volatile("global_load_async_to_lds_b128 %0, %1, off"
               :: "v"(lds_off), "v"((unsigned long long)(uintptr_t)gaddr)
               : "memory");
}

// conv1 A-gather offsets: off(K) = (K/9)*7056 + ((K%9)/3)*84 + (K%3), f32 elems.
// Slot i -> K = {0..7,16..23} for lane<16 (hi=0), {8..15,24..31} for lane>=16.
// K=27..31 are pad slots: B rows there are 0.0h, so any finite load is fine.
__constant__ const int OFF_LO[16] = {0, 1, 2, 84, 85, 86, 168, 169,
                                     7225, 7226, 14112, 14113, 14114, 14196, 14197, 14198};
__constant__ const int OFF_HI[16] = {170, 7056, 7057, 7058, 7140, 7141, 7142, 7224,
                                     14280, 14281, 14282, 14280, 14281, 14282, 14280, 14281};

// -------------------------------------------------------------------------
// Kernel 0: pre-swizzle fc_w [32][102400] f32 -> f16 in per-chunk WMMA-B
// lane layout: Bsw[cc][s:8][t:2][lane:32][i:16], cc = K/256.
// -------------------------------------------------------------------------
__global__ __launch_bounds__(256)
void k_prep(const float* __restrict__ fcw, _Float16* __restrict__ Bsw) {
  const int cc = blockIdx.x;
  const int tid = threadIdx.x;
  for (int idx = tid; idx < 8192; idx += 256) {
    const int s = idx >> 10, t = (idx >> 9) & 1, l = (idx >> 4) & 31, i = idx & 15;
    const int kl = s * 32 + (i & 7) + ((i >> 3) << 4) + ((l >> 4) << 3);
    const int j  = t * 16 + (l & 15);
    Bsw[(size_t)cc * 8192 + idx] =
        (_Float16)fcw[(size_t)j * FLAT_SZ + (size_t)cc * CHUNK + kl];
  }
}

// -------------------------------------------------------------------------
// Kernel 1: fully-WMMA fused conv1 + conv2, one workgroup per sample.
//  conv1: implicit GEMM M=1696(pad), N=32, K=27(pad 32), A gathered from x.
//  conv2: implicit GEMM M=1600, N=64, K=128; h1 lives in LDS as f16.
//  H2 written f16 in reference flatten order (c*1600 + pos).
// -------------------------------------------------------------------------
__global__ __launch_bounds__(256)
void k_conv_fused(const float* __restrict__ x,
                  const float* __restrict__ w1g, const float* __restrict__ b1g,
                  const float* __restrict__ w2g, const float* __restrict__ b2g,
                  _Float16* __restrict__ H2) {
  extern __shared__ char smem[];
  _Float16* h1  = (_Float16*)smem;              // [1696][32] f16 = 108544 B
  _Float16* Bs2 = (_Float16*)(smem + 108544);   // conv2 B: [4 ks][4 nt][32][16] = 16384 B
  _Float16* Bs1 = (_Float16*)(smem + 124928);   // conv1 B: [2 nt][32][16] = 2048 B
  float*    b1  = (float*)(smem + 126976);      // [32]
  float*    b2  = (float*)(smem + 127104);      // [64]

  const int tid = threadIdx.x;
  const int b   = blockIdx.x;
  const int lane = tid & 31, wave = tid >> 5;
  const int laneN = lane & 15, hi = lane >> 4;

  if (tid < 32) b1[tid] = b1g[tid];
  if (tid < 64) b2[tid] = b2g[tid];
  // conv1_w [32 o][27 k] -> B layout, K zero-padded 27->32
  for (int idx = tid; idx < 1024; idx += 256) {
    const int t = idx >> 9, l = (idx >> 4) & 31, i = idx & 15;
    const int K = (i & 7) + ((i >> 3) << 4) + ((l >> 4) << 3);
    const int o = t * 16 + (l & 15);
    Bs1[idx] = (_Float16)((K < 27) ? w1g[o * 27 + K] : 0.f);
  }
  // conv2_w [64 o][32 c][2 dy][2 dx] -> B layout per (kstep,ntile)
  for (int idx = tid; idx < 8192; idx += 256) {
    const int s = idx >> 11, t = (idx >> 9) & 3, l = (idx >> 4) & 31, i = idx & 15;
    const int c  = (i & 7) + ((i >> 3) << 4) + ((l >> 4) << 3);
    const int dy = s >> 1, dx = s & 1;
    const int o  = t * 16 + (l & 15);
    Bs2[idx] = (_Float16)w2g[((o * 32 + c) * 2 + dy) * 2 + dx];
  }
  __syncthreads();

  // ---- conv1 via WMMA: 106 full M-tiles, unconditional loads and stores ----
  const float* xb = x + (size_t)b * (3 * 84 * 84);
  for (int mt = wave; mt < 106; mt += 8) {
    const int posBase = mt * 16;
    const int pos = min(posBase + laneN, 1680);  // clamp load addresses only
    const int oy = pos / 41, ox = pos % 41;
    const float* pxb = xb + (oy * 2) * 84 + (ox * 2);
    v16h a;
#pragma unroll
    for (int i = 0; i < 16; ++i) {
      const int off = hi ? OFF_HI[i] : OFF_LO[i];  // one cndmask per slot
      a[i] = (_Float16)pxb[off];
    }
    v8f acc[2];
#pragma unroll
    for (int nt = 0; nt < 2; ++nt) {               // both WMMAs first (fills hazard slots)
      const float bias = b1[nt * 16 + laneN];
#pragma unroll
      for (int r = 0; r < 8; ++r) acc[nt][r] = bias;
      acc[nt] = wmma_f16(a, *(const v16h*)&Bs1[(nt * 32 + lane) * 16], acc[nt]);
    }
#pragma unroll
    for (int nt = 0; nt < 2; ++nt) {
#pragma unroll
      for (int r = 0; r < 8; ++r) {
        const int pm = posBase + hi * 8 + r;       // C row M = r + 8*hi, always < 1696
        h1[pm * 32 + nt * 16 + laneN] = relu_h((_Float16)acc[nt][r]);
      }
    }
  }
  __syncthreads();

  // ---- conv2 via WMMA: M=1600, N=64, K=128 (4 x k32), + relu + f16 store ----
  _Float16* outB = H2 + (size_t)b * FLAT_SZ;
  for (int mt = wave; mt < 100; mt += 8) {
    const int posBase = mt * 16;
    const int pos = posBase + laneN;
    const int oy = pos / 40, ox = pos % 40;
    v16h a[4];
#pragma unroll
    for (int s = 0; s < 4; ++s) {              // kstep = 2x2 tap: contiguous h1 row
      const int dy = s >> 1, dx = s & 1;
      const _Float16* row = &h1[((oy + dy) * 41 + (ox + dx)) * 32 + hi * 8];
      a[s] = cat16(*(const h8*)row, *(const h8*)(row + 16));
    }
    v8f acc[4];
#pragma unroll
    for (int nt = 0; nt < 4; ++nt) {           // 4 independent WMMA chains
      const float bias = b2[nt * 16 + laneN];
#pragma unroll
      for (int r = 0; r < 8; ++r) acc[nt][r] = bias;
#pragma unroll
      for (int s = 0; s < 4; ++s)
        acc[nt] = wmma_f16(a[s], *(const v16h*)&Bs2[(((s * 4 + nt) * 32) + lane) * 16],
                           acc[nt]);
    }
#pragma unroll
    for (int nt = 0; nt < 4; ++nt) {
      h8 st;
#pragma unroll
      for (int r = 0; r < 8; ++r) st[r] = relu_h((_Float16)acc[nt][r]);
      const int ch = nt * 16 + laneN;
      *(h8*)&outB[ch * 1600 + posBase + hi * 8] = st;   // contiguous 16B per lane
    }
  }
}

// -------------------------------------------------------------------------
// Kernel 2: fc GEMM [2048 x 102400] x [102400 x 32], WMMA f16, split-K x16.
// A and pre-swizzled B staged by global_load_async_to_lds_b128, double
// buffered with s_wait_asynccnt. Deterministic partials (no float atomics).
// -------------------------------------------------------------------------
__global__ __launch_bounds__(256)
void k_fc_gemm(const _Float16* __restrict__ H2, const _Float16* __restrict__ Bsw,
               float* __restrict__ partials) {
  extern __shared__ char smem2[];
  _Float16* As = (_Float16*)smem2;             // [2][16*256] = 16 KB
  _Float16* Bs = (_Float16*)(smem2 + 16384);   // [2][8192]   = 32 KB
  float*    red = (float*)(smem2 + 16384);     // [8][512] (reuses Bs after loop)

  const int tid = threadIdx.x;
  const int lane = tid & 31, wave = tid >> 5;
  const int laneN = lane & 15, hi = lane >> 4;
  const int rowBase = blockIdx.x * 16;
  const int ccBase  = blockIdx.y * NCHUNK;

  // Issue one chunk's staging: 6 async b128 instructions per wave.
  auto issue = [&](int c, int buf) {
    const int cc = ccBase + c;
    for (int i = tid; i < 512; i += 256) {           // A: 16 rows x 256 halfs
      const int r = i >> 5, kk = (i & 31) * 8;
      async_ld_b128((unsigned)(uintptr_t)&As[buf * 4096 + r * CHUNK + kk],
                    &H2[(size_t)(rowBase + r) * FLAT_SZ + (size_t)cc * CHUNK + kk]);
    }
    for (int i = tid; i < 1024; i += 256) {          // B: contiguous 16 KB
      async_ld_b128((unsigned)(uintptr_t)&Bs[buf * 8192 + i * 8],
                    &Bsw[(size_t)cc * 8192 + i * 8]);
    }
  };

  v8f acc0 = {}, acc1 = {};
  issue(0, 0);
  for (int c = 0; c < NCHUNK; ++c) {
    const int buf = c & 1;
    if (c + 1 < NCHUNK) {
      issue(c + 1, buf ^ 1);                         // overlap next chunk's DMA
      asm volatile("s_wait_asynccnt 0x6" ::: "memory");
    } else {
      asm volatile("s_wait_asynccnt 0x0" ::: "memory");
    }
    __syncthreads();
    const _Float16* ab = &As[buf * 4096 + laneN * CHUNK + wave * 32 + hi * 8];
    const v16h a = cat16(*(const h8*)ab, *(const h8*)(ab + 16));
    acc0 = wmma_f16(a, *(const v16h*)&Bs[buf * 8192 + wave * 1024 + lane * 16], acc0);
    acc1 = wmma_f16(a, *(const v16h*)&Bs[buf * 8192 + wave * 1024 + 512 + lane * 16], acc1);
    __syncthreads();
  }

  // cross-wave reduction (red overlaps Bs; loop's trailing barrier protects it)
#pragma unroll
  for (int r = 0; r < 8; ++r) {
    red[wave * 512 + (r + hi * 8) * 32 + laneN]      = acc0[r];
    red[wave * 512 + (r + hi * 8) * 32 + 16 + laneN] = acc1[r];
  }
  __syncthreads();
  for (int i = tid; i < 512; i += 256) {
    float v = 0.f;
#pragma unroll
    for (int w = 0; w < 8; ++w) v += red[w * 512 + i];
    partials[((size_t)blockIdx.y * B_SZ + rowBase + (i >> 5)) * 32 + (i & 31)] = v;
  }
}

// -------------------------------------------------------------------------
// Kernel 3: split-K reduce + fc bias/relu, concat y, cb1/cb2/relu, routed head.
// -------------------------------------------------------------------------
__global__ __launch_bounds__(256)
void k_tail(const float* __restrict__ partials, const float* __restrict__ y,
            const int* __restrict__ head, const float* __restrict__ fcb,
            const float* __restrict__ cb1w, const float* __restrict__ cb1b,
            const float* __restrict__ cb2w, const float* __restrict__ cb2b,
            const float* __restrict__ hw, const float* __restrict__ hb,
            float* __restrict__ out) {
  __shared__ float s_cb1w[32 * 36], s_cb2w[12 * 32], s_hw[8 * 4 * 12];
  __shared__ float s_cb1b[32], s_cb2b[12], s_hb[32], s_fcb[32];
  const int tid = threadIdx.x;
  for (int i = tid; i < 1152; i += 256) s_cb1w[i] = cb1w[i];
  for (int i = tid; i < 384;  i += 256) s_cb2w[i] = cb2w[i];
  for (int i = tid; i < 384;  i += 256) s_hw[i]   = hw[i];
  if (tid < 32) { s_cb1b[tid] = cb1b[tid]; s_fcb[tid] = fcb[tid]; s_hb[tid] = hb[tid]; }
  if (tid < 12) s_cb2b[tid] = cb2b[tid];
  __syncthreads();

  const int b = blockIdx.x * 256 + tid;
  float c[36];
#pragma unroll 4
  for (int j = 0; j < 32; ++j) {
    float s = s_fcb[j];
    for (int ks = 0; ks < KSLICES; ++ks)
      s += partials[((size_t)ks * B_SZ + b) * 32 + j];
    c[j] = fmaxf(s, 0.f);
  }
#pragma unroll
  for (int j = 0; j < 4; ++j) c[32 + j] = y[(size_t)b * 4 + j];

  float r1[32];
#pragma unroll 2
  for (int o = 0; o < 32; ++o) {
    float s = s_cb1b[o];
#pragma unroll
    for (int k = 0; k < 36; ++k) s = fmaf(c[k], s_cb1w[o * 36 + k], s);
    r1[o] = fmaxf(s, 0.f);
  }
  float r2[12];
#pragma unroll
  for (int o = 0; o < 12; ++o) {
    float s = s_cb2b[o];
#pragma unroll
    for (int k = 0; k < 32; ++k) s = fmaf(r1[k], s_cb2w[o * 32 + k], s);
    r2[o] = fmaxf(s, 0.f);
  }
  int idx = head[b]; if (idx > 7) idx = 7;
#pragma unroll
  for (int a = 0; a < 4; ++a) {
    float s = s_hb[idx * 4 + a];
#pragma unroll
    for (int k = 0; k < 12; ++k) s = fmaf(r2[k], s_hw[(idx * 4 + a) * 12 + k], s);
    out[(size_t)b * 4 + a] = s;
  }
}

// -------------------------------------------------------------------------
extern "C" void kernel_launch(void* const* d_in, const int* in_sizes, int n_in,
                              void* d_out, int out_size, void* d_ws, size_t ws_size,
                              hipStream_t stream) {
  const float* x    = (const float*)d_in[0];
  const float* y    = (const float*)d_in[1];
  const int*   head = (const int*)d_in[2];
  const float* c1w  = (const float*)d_in[3];
  const float* c1b  = (const float*)d_in[4];
  const float* c2w  = (const float*)d_in[5];
  const float* c2b  = (const float*)d_in[6];
  const float* fcw  = (const float*)d_in[7];
  const float* fcb  = (const float*)d_in[8];
  const float* cb1w = (const float*)d_in[9];
  const float* cb1b = (const float*)d_in[10];
  const float* cb2w = (const float*)d_in[11];
  const float* cb2b = (const float*)d_in[12];
  const float* hw   = (const float*)d_in[13];
  const float* hb   = (const float*)d_in[14];
  float* out = (float*)d_out;

  // workspace: H2 f16 (419.4 MB) | partials f32 (4.2 MB) | Bsw f16 (6.5 MB)
  _Float16* H2 = (_Float16*)d_ws;
  float* partials = (float*)((char*)d_ws + (size_t)B_SZ * FLAT_SZ * sizeof(_Float16));
  _Float16* Bsw = (_Float16*)((char*)partials +
                              (size_t)KSLICES * B_SZ * 32 * sizeof(float));

  const int lds1 = 127360;  // h1(108544) + Bs2 + Bs1 + biases
  const int lds2 = 49152;   // As[2] + Bs[2] (red overlaps Bs)
  hipFuncSetAttribute(reinterpret_cast<const void*>(k_conv_fused),
                      hipFuncAttributeMaxDynamicSharedMemorySize, lds1);
  hipFuncSetAttribute(reinterpret_cast<const void*>(k_fc_gemm),
                      hipFuncAttributeMaxDynamicSharedMemorySize, lds2);

  k_prep<<<NCHUNK_TOT, 256, 0, stream>>>(fcw, Bsw);
  k_conv_fused<<<B_SZ, 256, lds1, stream>>>(x, c1w, c1b, c2w, c2b, H2);
  k_fc_gemm<<<dim3(B_SZ / 16, KSLICES), 256, lds2, stream>>>(H2, Bsw, partials);
  k_tail<<<B_SZ / 256, 256, 0, stream>>>(partials, y, head, fcb,
                                         cb1w, cb1b, cb2w, cb2b, hw, hb, out);
}
